// NgramMultiheadAttention_19516331393551
// MI455X (gfx1250) — matrix operand order repaired and checked by reading, not compiled, not run on hardware
//
#include <hip/hip_runtime.h>
#include <hip/hip_bf16.h>

// Problem constants
#define E_   1024
#define H_   16
#define NG_  2
#define TM_  512
#define B_   4
#define NB_  32
#define HD_  64
#define T3_  1536   // (NG_+1)*TM_
#define BH_  64     // B_*H_

typedef __attribute__((ext_vector_type(16))) __bf16 v16bf;
typedef __attribute__((ext_vector_type(8)))  float  v8f;

// Native conversions -> backend emits v_cvt_pk_bf16_f32 / v_cvt_bf16_f32
static __device__ __forceinline__ unsigned short f2bf(float a) {
    return __builtin_bit_cast(unsigned short, (__bf16)a);
}
static __device__ __forceinline__ unsigned int pack2(float a, float b) {
    return (unsigned int)f2bf(a) | ((unsigned int)f2bf(b) << 16);
}

union Frag16 { unsigned int u[8]; v16bf v; };

// ---------------------------------------------------------------------------
// One-time f32 -> bf16 elementwise conversion (float4 in, 4xbf16 out)
// ---------------------------------------------------------------------------
__global__ void __launch_bounds__(256)
cvt_f32_bf16(const float* __restrict__ in, unsigned short* __restrict__ out, int n4)
{
    const int i = blockIdx.x * 256 + threadIdx.x;
    if (i >= n4) return;
    const float4 f = ((const float4*)in)[i];
    uint2 p;
    p.x = pack2(f.x, f.y);
    p.y = pack2(f.z, f.w);
    ((uint2*)out)[i] = p;
}

// ---------------------------------------------------------------------------
// WMMA GEMM on pre-converted bf16:  out = (A @ W^T + bias) * scale
//   A: (M,K) bf16 row-major;  Wbf: (N,K) bf16 row-major
//   One wave computes a 32(M) x 64(N) strip; block = 4 waves.
//   Inner loop is pure b128 loads + v_wmma (no conversions).
//   mode 0: bf16 out at (bh, t, d)      [q / k]
//   mode 1: bf16 out at (bh, d, t)      [v transposed]
//   mode 2: f32  out at (bh, t, nb)     [relative-bucket vals]
//   mode 3: f32  out row-major (M,N)    [final output projection]
// ---------------------------------------------------------------------------
__global__ void __launch_bounds__(128)
gemm_bf16_wmma(const unsigned short* __restrict__ Abf,
               const unsigned short* __restrict__ Wbf,
               const float* __restrict__ bias,
               int M, int N, int K, float scale, int mode,
               unsigned short* __restrict__ obf,
               float* __restrict__ of32)
{
    const int lane = threadIdx.x & 31;
    const int wave = threadIdx.x >> 5;
    const int nStrips = N >> 6;
    const int strip = blockIdx.x * 4 + wave;
    if (strip >= (M >> 5) * nStrips) return;
    const int tm = strip / nStrips;          // 32-row tile
    const int tn = strip % nStrips;          // 64-col tile

    const int nlo = lane & 15;
    const int ha  = lane >> 4;

    const v8f vz = {0.f,0.f,0.f,0.f,0.f,0.f,0.f,0.f};
    v8f acc[2][4] = {{vz, vz, vz, vz}, {vz, vz, vz, vz}};

    // Per-lane base pointers (A rows for the two 16-row subtiles, B cols)
    const unsigned short* arow0 = Abf + (size_t)(tm * 32 + nlo)      * K;
    const unsigned short* arow1 = Abf + (size_t)(tm * 32 + 16 + nlo) * K;
    const unsigned short* wrow  = Wbf + (size_t)(tn * 64 + nlo) * K + 16 * ha;

    for (int k0 = 0; k0 < K; k0 += 32) {
        if (k0 + 32 < K) {
            __builtin_prefetch(arow0 + k0 + 32, 0, 3);   // WGP-scope prefetch
            __builtin_prefetch(wrow + k0 + 32, 0, 3);
        }
        // ---- two A fragments (16x32 each)
        Frag16 a[2];
#pragma unroll
        for (int v = 0; v < 8; ++v) {
            const int koff = k0 + 16 * ha + 2 * v;
            a[0].u[v] = *(const unsigned int*)(arow0 + koff);
            a[1].u[v] = *(const unsigned int*)(arow1 + koff);
        }
        // ---- 4 B fragments; each feeds 2 WMMAs
#pragma unroll
        for (int nt = 0; nt < 4; ++nt) {
            const unsigned short* wr = wrow + (size_t)nt * 16 * K + k0;
            Frag16 bfr;
#pragma unroll
            for (int v = 0; v < 8; ++v)
                bfr.u[v] = *(const unsigned int*)(wr + 2 * v);
#pragma unroll
            for (int mt = 0; mt < 2; ++mt)
                acc[mt][nt] = __builtin_amdgcn_wmma_f32_16x16x32_bf16(
                    false, a[mt].v, false, bfr.v, (short)0, acc[mt][nt], false, false);
        }
    }

    // ---- epilogue: C layout — lane(l): col = l&15, row = r + 8*(l>>4)
#pragma unroll
    for (int mt = 0; mt < 2; ++mt)
#pragma unroll
    for (int nt = 0; nt < 4; ++nt) {
        const int ncol = tn * 64 + nt * 16 + nlo;
        const float bv = bias ? bias[ncol] : 0.f;
#pragma unroll
        for (int r = 0; r < 8; ++r) {
            const int row = tm * 32 + mt * 16 + r + 8 * ha;
            const float val = (acc[mt][nt][r] + bv) * scale;
            const int t = row / B_, b = row % B_;   // row index m = t*B + b
            if (mode == 0) {
                const int h = ncol >> 6, d = ncol & 63;
                obf[(((size_t)(b * H_ + h) * T3_ + t) << 6) + d] = f2bf(val);
            } else if (mode == 1) {
                const int h = ncol >> 6, d = ncol & 63;
                obf[((size_t)((b * H_ + h) * HD_ + d)) * T3_ + t] = f2bf(val);
            } else if (mode == 2) {
                const int h = ncol & 15, nb = ncol >> 4;   // c = nb*H + h
                of32[((size_t)(b * H_ + h) * T3_ + t) * NB_ + nb] = val;
            } else {
                of32[(size_t)row * N + ncol] = val;
            }
        }
    }
}

// ---------------------------------------------------------------------------
// Flash attention over 3 streams (main + 2 ngram).
// grid = (8, 64, 3); block = 128 (4 waves); each wave owns 16 queries.
// K / V^T tiles for the 32-key step are staged in LDS once per block and
// shared by all 4 waves (4x VMEM reduction).
// ---------------------------------------------------------------------------
__global__ void __launch_bounds__(128)
flash_ngram(const unsigned short* __restrict__ qb,   // (bh, T3, 64) bf16
            const unsigned short* __restrict__ kb,   // (bh, T3, 64) bf16
            const unsigned short* __restrict__ vtb,  // (bh, 64, T3) bf16
            const float* __restrict__ rvals,         // (bh, T3, 32) f32
            const float* __restrict__ smask,         // (512, 512)
            const float* __restrict__ ngmask,        // (2, 512, 1024)
            const int* __restrict__ ibm,             // (4, 512, 512)
            const int* __restrict__ ibr,             // (4, 512, 1024)
            unsigned short* __restrict__ ctx)        // (T3, B, E) bf16
{
    __shared__ unsigned short kst[32][72];           // K tile: key x d   (pad->16B rows)
    __shared__ unsigned short vst[64][40];           // V^T tile: d x key (pad->16B rows)
    __shared__ unsigned short pst[4][16][34];        // per-wave P staging

    const int tid  = threadIdx.x;
    const int lane = tid & 31;
    const int wave = tid >> 5;
    const int nlo  = lane & 15;
    const int ha   = lane >> 4;

    const int s  = blockIdx.z;
    const int bh = blockIdx.y;
    const int b  = bh >> 4, h = bh & 15;             // bh = b*H + h
    const int qt0 = blockIdx.x * 64 + wave * 16;     // query base within stream
    const int Slen = (s == 0) ? TM_ : 2 * TM_;

    // ---- preload Q fragments (16 queries x 64 d -> two K=32 A-frags)
    Frag16 qf[2];
    {
        const int t3q = s * TM_ + qt0 + nlo;
        const unsigned short* qr = qb + ((size_t)bh * T3_ + t3q) * HD_;
#pragma unroll
        for (int f = 0; f < 2; ++f)
#pragma unroll
        for (int v = 0; v < 8; ++v) {
            const int koff = f * 32 + 16 * ha + 2 * v;
            qf[f].u[v] = *(const unsigned int*)(qr + koff);
        }
    }

    const v8f vz = {0.f,0.f,0.f,0.f,0.f,0.f,0.f,0.f};
    v8f O[4] = {vz, vz, vz, vz};
    float mrun[8], lrun[8];
#pragma unroll
    for (int r = 0; r < 8; ++r) { mrun[r] = -3.0e38f; lrun[r] = 0.f; }

    for (int j0 = 0; j0 < Slen; j0 += 32) {
        // key rows: main region [0,512) else this stream's region
        const int t3k0 = (s > 0 && j0 >= TM_) ? (s * TM_ + j0 - TM_) : j0;

        // ---- cooperative stage of K (32x64) and V^T (64x32) tiles into LDS
        __syncthreads();                              // previous tile fully consumed
        {
            const unsigned short* ksrc = kb + ((size_t)bh * T3_ + t3k0) * HD_;
            const unsigned short* vsrc = vtb + (size_t)bh * HD_ * T3_ + t3k0;
#pragma unroll
            for (int it = 0; it < 2; ++it) {
                const int e = (tid + it * 128) * 8;   // 8 bf16 per b128
                const int kj = e >> 6, kd = e & 63;   // K tile coords
                *(uint4*)&kst[kj][kd] = *(const uint4*)(ksrc + (size_t)kj * HD_ + kd);
                const int vd = e >> 5, vc = e & 31;   // V tile coords
                *(uint4*)&vst[vd][vc] = *(const uint4*)(vsrc + (size_t)vd * T3_ + vc);
            }
        }
        __syncthreads();

        // ---- S = Q @ K^T  (two 16x16 key subtiles), K from LDS
        v8f Sf[2] = {vz, vz};
#pragma unroll
        for (int jj = 0; jj < 2; ++jj) {
            const unsigned short* kr = &kst[jj * 16 + nlo][0];
#pragma unroll
            for (int f = 0; f < 2; ++f) {
                Frag16 bfr;
#pragma unroll
                for (int v = 0; v < 8; ++v) {
                    const int d = f * 32 + 16 * ha + 2 * v;
                    bfr.u[v] = *(const unsigned int*)(kr + d);
                }
                Sf[jj] = __builtin_amdgcn_wmma_f32_16x16x32_bf16(
                    false, qf[f].v, false, bfr.v, (short)0, Sf[jj], false, false);
            }
        }

        // ---- add relative-bucket logits (gather) + mask
#pragma unroll
        for (int jj = 0; jj < 2; ++jj) {
            const int j = j0 + jj * 16 + nlo;
#pragma unroll
            for (int r = 0; r < 8; ++r) {
                const int tl = qt0 + r + 8 * ha;
                float mk; int idx;
                if (s == 0) {
                    mk  = smask[(size_t)tl * TM_ + j];
                    idx = ibm[((size_t)b * TM_ + tl) * TM_ + j];
                } else {
                    mk  = ngmask[((size_t)(s - 1) * TM_ + tl) * (2 * TM_) + j];
                    idx = ibr[((size_t)b * TM_ + tl) * (2 * TM_) + j];
                }
                const float bv =
                    rvals[((size_t)bh * T3_ + s * TM_ + tl) * NB_ + idx];
                Sf[jj][r] += mk + bv;
            }
        }

        // ---- online softmax (row stats shared across the 16-lane half)
#pragma unroll
        for (int r = 0; r < 8; ++r) {
            float x = fmaxf(Sf[0][r], Sf[1][r]);
            x = fmaxf(x, __shfl_xor(x, 1, 32));
            x = fmaxf(x, __shfl_xor(x, 2, 32));
            x = fmaxf(x, __shfl_xor(x, 4, 32));
            x = fmaxf(x, __shfl_xor(x, 8, 32));
            const float mnew  = fmaxf(mrun[r], x);
            const float alpha = __expf(mrun[r] - mnew);
            mrun[r] = mnew;
            const float p0 = __expf(Sf[0][r] - mnew);
            const float p1 = __expf(Sf[1][r] - mnew);
            Sf[0][r] = p0; Sf[1][r] = p1;
            float rs = p0 + p1;
            rs += __shfl_xor(rs, 1, 32);
            rs += __shfl_xor(rs, 2, 32);
            rs += __shfl_xor(rs, 4, 32);
            rs += __shfl_xor(rs, 8, 32);
            lrun[r] = lrun[r] * alpha + rs;
#pragma unroll
            for (int nt = 0; nt < 4; ++nt) O[nt][r] *= alpha;
        }

        // ---- P: C-layout -> bf16 -> LDS -> A-layout fragment (wave-private)
#pragma unroll
        for (int jj = 0; jj < 2; ++jj)
#pragma unroll
        for (int r = 0; r < 8; ++r)
            pst[wave][r + 8 * ha][jj * 16 + nlo] = f2bf(Sf[jj][r]);

        Frag16 pf;
#pragma unroll
        for (int v = 0; v < 8; ++v) {
            const int koff = 16 * ha + 2 * v;
            pf.u[v] = *(const unsigned int*)&pst[wave][nlo][koff];
        }

        // ---- O += P @ V, V^T from LDS (key pairs contiguous)
#pragma unroll
        for (int nt = 0; nt < 4; ++nt) {
            const unsigned short* vr = &vst[nt * 16 + nlo][16 * ha];
            Frag16 bfr;
#pragma unroll
            for (int v = 0; v < 8; ++v)
                bfr.u[v] = *(const unsigned int*)(vr + 2 * v);
            O[nt] = __builtin_amdgcn_wmma_f32_16x16x32_bf16(
                false, pf.v, false, bfr.v, (short)0, O[nt], false, false);
        }
    }

    // ---- normalize and store context (t, b, e) bf16
#pragma unroll
    for (int nt = 0; nt < 4; ++nt)
#pragma unroll
    for (int r = 0; r < 8; ++r) {
        const int tl = qt0 + r + 8 * ha;
        const int d  = nt * 16 + nlo;
        const float val = O[nt][r] / lrun[r];
        ctx[((size_t)(s * TM_ + tl) * B_ + b) * E_ + h * HD_ + d] = f2bf(val);
    }
}

// ---------------------------------------------------------------------------
extern "C" void kernel_launch(void* const* d_in, const int* in_sizes, int n_in,
                              void* d_out, int out_size, void* d_ws, size_t ws_size,
                              hipStream_t stream) {
    const float* hs  = (const float*)d_in[0];
    const float* sm  = (const float*)d_in[1];
    const float* ngm = (const float*)d_in[2];
    const int*   ibm = (const int*)d_in[3];
    const int*   ibr = (const int*)d_in[4];
    const float* Wq = (const float*)d_in[5];  const float* bq = (const float*)d_in[6];
    const float* Wk = (const float*)d_in[7];  const float* bk = (const float*)d_in[8];
    const float* Wv = (const float*)d_in[9];  const float* bv = (const float*)d_in[10];
    const float* Wo = (const float*)d_in[11]; const float* bo = (const float*)d_in[12];
    const float* Wr = (const float*)d_in[13]; const float* br = (const float*)d_in[14];

    char* ws = (char*)d_ws;
    auto carve = [&](size_t bytes) { void* p = ws; ws += bytes; return p; };

    const size_t sz_qkv = (size_t)BH_ * T3_ * HD_ * 2;          // 12.6MB each
    unsigned short* qbuf = (unsigned short*)carve(sz_qkv);
    unsigned short* kbuf = (unsigned short*)carve(sz_qkv);
    unsigned short* vbuf = (unsigned short*)carve(sz_qkv);
    float* rvals = (float*)carve((size_t)BH_ * T3_ * NB_ * 4);
    unsigned short* ctx  = (unsigned short*)carve((size_t)T3_ * B_ * E_ * 2);
    // bf16 copies of activations + weights
    unsigned short* hsb  = (unsigned short*)carve((size_t)T3_ * B_ * E_ * 2);
    unsigned short* wqb  = (unsigned short*)carve((size_t)E_ * E_ * 2);
    unsigned short* wkb  = (unsigned short*)carve((size_t)E_ * E_ * 2);
    unsigned short* wvb  = (unsigned short*)carve((size_t)E_ * E_ * 2);
    unsigned short* wob  = (unsigned short*)carve((size_t)E_ * E_ * 2);
    unsigned short* wrb  = (unsigned short*)carve((size_t)NB_ * H_ * E_ * 2);

    const int M = T3_ * B_;   // 6144
    const dim3 blk128(128), blk256(256);

    // ---- one-time f32 -> bf16 conversions (pure bandwidth, ~50MB)
    auto cvt = [&](const float* src, unsigned short* dst, size_t n) {
        const int n4 = (int)(n / 4);
        cvt_f32_bf16<<<(n4 + 255) / 256, blk256, 0, stream>>>(src, dst, n4);
    };
    cvt(hs, hsb, (size_t)M * E_);
    cvt(Wq, wqb, (size_t)E_ * E_);
    cvt(Wk, wkb, (size_t)E_ * E_);
    cvt(Wv, wvb, (size_t)E_ * E_);
    cvt(Wo, wob, (size_t)E_ * E_);
    cvt(Wr, wrb, (size_t)NB_ * H_ * E_);

    auto nblocks = [](int m, int n) { return ((m / 32) * (n / 64) + 3) / 4; };

    // Q (pre-scaled by 1/sqrt(64)), K, V^T, relative-bucket vals
    gemm_bf16_wmma<<<nblocks(M, E_), blk128, 0, stream>>>(
        hsb, wqb, bq, M, E_, E_, 0.125f, 0, qbuf, nullptr);
    gemm_bf16_wmma<<<nblocks(M, E_), blk128, 0, stream>>>(
        hsb, wkb, bk, M, E_, E_, 1.0f, 0, kbuf, nullptr);
    gemm_bf16_wmma<<<nblocks(M, E_), blk128, 0, stream>>>(
        hsb, wvb, bv, M, E_, E_, 1.0f, 1, vbuf, nullptr);
    gemm_bf16_wmma<<<nblocks(M, NB_ * H_), blk128, 0, stream>>>(
        hsb, wrb, br, M, NB_ * H_, E_, 1.0f, 2, nullptr, rvals);

    // fused gather + mask + softmax + P@V, all three streams
    flash_ngram<<<dim3(TM_ / 64, BH_, NG_ + 1), blk128, 0, stream>>>(
        qbuf, kbuf, vbuf, rvals, sm, ngm, ibm, ibr, ctx);

    // output projection straight into d_out (f32, (T3,B,E))
    gemm_bf16_wmma<<<nblocks(M, E_), blk128, 0, stream>>>(
        ctx, wob, bo, M, E_, E_, 1.0f, 3, nullptr, (float*)d_out);
}
